// CharRNN_33517924778348
// MI455X (gfx1250) — matrix operand order repaired and checked by reading, not compiled, not run on hardware
//
#include <hip/hip_runtime.h>
#include <hip/hip_bf16.h>
#include <math.h>

typedef __attribute__((ext_vector_type(16))) __bf16 v16bf;
typedef __attribute__((ext_vector_type(8)))  __bf16 v8bf;
typedef __attribute__((ext_vector_type(8)))  float  v8f;

#define B_    64
#define T_    128
#define U_    256
#define V_    16000
#define G4    1024     // 4*U
#define K2    512      // 2*U
#define NROWS 8192     // B*T

__device__ __forceinline__ float sigmoid_f(float x) { return 1.0f / (1.0f + __expf(-x)); }

// A fragment (16x32 bf16), contiguous k order (same as load_b): row = mbase + (lane&15).
// lanes 0-15 hold K = k0..k0+15; lanes 16-31 K = k0+16..k0+31.
__device__ __forceinline__ v16bf load_a(const __bf16* A, int row, int ldk, int k0, int laneHi) {
    const __bf16* p = A + row * ldk + k0 + (laneHi ? 16 : 0);
    v8bf lo = *(const v8bf*)(p);
    v8bf hi = *(const v8bf*)(p + 8);
    v16bf a;
#pragma unroll
    for (int i = 0; i < 8; ++i) { a[i] = lo[i]; a[8 + i] = hi[i]; }
    return a;
}

// B fragment (32x16 bf16) from K-major weights Wt[n*ldk + k]:
// col n = nbase + (lane&15); lanes 0-15 hold K=k0..k0+15, lanes 16-31 K=k0+16..k0+31.
// One contiguous 32-byte load per lane.
__device__ __forceinline__ v16bf load_b(const __bf16* Wt, int n, int ldk, int k0, int laneHi) {
    return *(const v16bf*)(Wt + n * ldk + k0 + (laneHi ? 16 : 0));
}

__device__ __forceinline__ v8f wmma_bf16(v16bf a, v16bf b, v8f c) {
    return __builtin_amdgcn_wmma_f32_16x16x32_bf16(false, a, false, b, (short)0, c, false, false);
}

// ---------------- conversion / gather kernels ----------------

// float [K,N] row-major -> bf16 [N,K] (K-major for B fragments)
__global__ void k_f2bfT(const float* __restrict__ W, __bf16* __restrict__ Wt, int K, int N) {
    int idx = blockIdx.x * 256 + threadIdx.x;
    if (idx >= K * N) return;
    int k = idx / N, n = idx - k * N;
    Wt[n * K + k] = (__bf16)W[idx];
}

// embedding gather -> time-major bf16 x[t*B+b][U]
__global__ void k_embed(const int* __restrict__ ids, const float* __restrict__ E,
                        __bf16* __restrict__ Xbf) {
    int tb = blockIdx.x;               // t*64 + b
    int t = tb >> 6, b = tb & 63;
    int id = ids[b * T_ + t];
    int u = threadIdx.x;
    Xbf[tb * U_ + u] = (__bf16)E[id * U_ + u];
}

// ---------------- Zx0 = X @ W0x + b0   ([8192,256]x[256,1024]) ----------------
// grid (128, 32), block 256 = 8 waves (4 M-tiles x 2 N-tiles)
__global__ void k_zx(const __bf16* __restrict__ X, const __bf16* __restrict__ W0t,
                     const float* __restrict__ b0, float* __restrict__ Z) {
    const int lane = threadIdx.x & 31, wave = threadIdx.x >> 5;
    const int laneHi = lane >> 4, ln = lane & 15;
    const int mi = wave >> 1, ni = wave & 1;
    const int mbase = blockIdx.x * 64 + mi * 16;
    const int nbase = blockIdx.y * 32 + ni * 16;

    v8f acc = {};
#pragma unroll
    for (int k0 = 0; k0 < U_; k0 += 32) {
        v16bf a = load_a(X, mbase + ln, U_, k0, laneHi);
        v16bf b = load_b(W0t, nbase + ln, K2, k0, laneHi);   // x-part: k in [0,256)
        acc = wmma_bf16(a, b, acc);
    }
#pragma unroll
    for (int v = 0; v < 8; ++v) {
        int row = mbase + v + 8 * laneHi;
        int col = nbase + ln;
        Z[row * G4 + col] = acc[v] + b0[col];
    }
}

// ---------------- LSTM layer 0 step: z = Zx_t + h_prev @ W0h ----------------
// grid 8, block 256: 64 waves total; wave -> (mi 0..3, gate-col tile ci 0..15)
__global__ void k_lstm_l0(const float* __restrict__ Zx_t,      // [64,1024]
                          const __bf16* __restrict__ hprev,    // [64,256]
                          const __bf16* __restrict__ W0t,      // [1024,512] K-major
                          float* __restrict__ c,               // [64,256] in/out
                          __bf16* __restrict__ hout) {         // [64,256]
    const int lane = threadIdx.x & 31, wave = threadIdx.x >> 5;
    const int laneHi = lane >> 4, ln = lane & 15;
    const int gw = blockIdx.x * 8 + wave;
    const int mbase = (gw >> 4) * 16;
    const int cbase = (gw & 15) * 16;

    v8f acc[4];
#pragma unroll
    for (int g = 0; g < 4; ++g) {
        const int col = g * U_ + cbase + ln;
#pragma unroll
        for (int v = 0; v < 8; ++v)
            acc[g][v] = Zx_t[(mbase + v + 8 * laneHi) * G4 + col];
    }
#pragma unroll
    for (int k0 = 0; k0 < U_; k0 += 32) {
        v16bf a = load_a(hprev, mbase + ln, U_, k0, laneHi);
#pragma unroll
        for (int g = 0; g < 4; ++g) {
            v16bf b = load_b(W0t, g * U_ + cbase + ln, K2, U_ + k0, laneHi); // h-part
            acc[g] = wmma_bf16(a, b, acc[g]);
        }
    }
#pragma unroll
    for (int v = 0; v < 8; ++v) {
        int row = mbase + v + 8 * laneHi;
        int col = cbase + ln;
        float iv = acc[0][v], jv = acc[1][v], fv = acc[2][v], ov = acc[3][v];
        float cp = c[row * U_ + col];
        float cn = cp * sigmoid_f(fv + 1.0f) + sigmoid_f(iv) * tanhf(jv);
        c[row * U_ + col] = cn;
        hout[row * U_ + col] = (__bf16)(tanhf(cn) * sigmoid_f(ov));
    }
}

// ---------------- LSTM layer 1 step: z = b1 + [h0,h1] @ W1 ----------------
__global__ void k_lstm_l1(const float* __restrict__ b1,
                          const __bf16* __restrict__ h0cur,    // [64,256]
                          const __bf16* __restrict__ h1prev,   // [64,256]
                          const __bf16* __restrict__ W1t,      // [1024,512] K-major
                          float* __restrict__ c,               // [64,256] in/out
                          __bf16* __restrict__ hout,           // [64,256]
                          __bf16* __restrict__ outrow) {       // [64,256] time slice of outs
    const int lane = threadIdx.x & 31, wave = threadIdx.x >> 5;
    const int laneHi = lane >> 4, ln = lane & 15;
    const int gw = blockIdx.x * 8 + wave;
    const int mbase = (gw >> 4) * 16;
    const int cbase = (gw & 15) * 16;

    v8f acc[4];
#pragma unroll
    for (int g = 0; g < 4; ++g) {
        float bias = b1[g * U_ + cbase + ln];
#pragma unroll
        for (int v = 0; v < 8; ++v) acc[g][v] = bias;
    }
#pragma unroll
    for (int k0 = 0; k0 < K2; k0 += 32) {
        const __bf16* Asrc = (k0 < U_) ? h0cur : h1prev;
        const int kk = (k0 < U_) ? k0 : (k0 - U_);
        v16bf a = load_a(Asrc, mbase + ln, U_, kk, laneHi);
#pragma unroll
        for (int g = 0; g < 4; ++g) {
            v16bf b = load_b(W1t, g * U_ + cbase + ln, K2, k0, laneHi);
            acc[g] = wmma_bf16(a, b, acc[g]);
        }
    }
#pragma unroll
    for (int v = 0; v < 8; ++v) {
        int row = mbase + v + 8 * laneHi;
        int col = cbase + ln;
        float iv = acc[0][v], jv = acc[1][v], fv = acc[2][v], ov = acc[3][v];
        float cp = c[row * U_ + col];
        float cn = cp * sigmoid_f(fv + 1.0f) + sigmoid_f(iv) * tanhf(jv);
        float hn = tanhf(cn) * sigmoid_f(ov);
        c[row * U_ + col] = cn;
        hout[row * U_ + col] = (__bf16)hn;
        outrow[row * U_ + col] = (__bf16)hn;
    }
}

// ------------- fused projection + log-softmax + target gather -------------
// grid 128 (64 rows each), block 256 = 8 waves (4 M-tiles x 2 N-tiles)
__global__ void k_proj_nll(const __bf16* __restrict__ H,    // [8192,256] time-major
                           const __bf16* __restrict__ SWt,  // [16000,256] K-major
                           const float* __restrict__ sb,    // [16000]
                           const int* __restrict__ targets, // [64,128] (B,T)
                           float* __restrict__ nll) {       // [8192]
    __shared__ float s_m[64][33];
    __shared__ float s_s[64][33];
    __shared__ float s_tgt[64];
    __shared__ int   s_tc[64];

    const int tid = threadIdx.x;
    const int lane = tid & 31, wave = tid >> 5;
    const int laneHi = lane >> 4, ln = lane & 15;
    const int mi = wave >> 1, ni = wave & 1;
    const int rbase = blockIdx.x * 64;
    const int mbase = mi * 16;

    if (tid < 64) {
        int gr = rbase + tid;           // time-major: gr = t*64 + b
        int t = gr >> 6, b = gr & 63;
        s_tc[tid] = targets[b * T_ + t];
    }
    __syncthreads();

    // rows are fixed: keep the whole K=256 A operand in registers (8 fragments)
    v16bf afrag[8];
#pragma unroll
    for (int kk = 0; kk < 8; ++kk)
        afrag[kk] = load_a(H + rbase * U_, mbase + ln, U_, kk * 32, laneHi);

    float rm[8], rs[8], tv[8];
    bool cap[8];
#pragma unroll
    for (int v = 0; v < 8; ++v) { rm[v] = -INFINITY; rs[v] = 0.0f; tv[v] = 0.0f; cap[v] = false; }

    for (int it = 0; it < V_ / 32; ++it) {
        const int col = it * 32 + ni * 16 + ln;
        if (col + 32 < V_) __builtin_prefetch(SWt + (size_t)(col + 32) * U_, 0, 0);
        v8f acc = {};
#pragma unroll
        for (int kk = 0; kk < 8; ++kk) {
            v16bf b = load_b(SWt, col, U_, kk * 32, laneHi);
            acc = wmma_bf16(afrag[kk], b, acc);
        }
        float bias = sb[col];
#pragma unroll
        for (int v = 0; v < 8; ++v) {
            float x = acc[v] + bias;
            int lrow = mbase + v + 8 * laneHi;
            if (col == s_tc[lrow]) { tv[v] = x; cap[v] = true; }
            float m = fmaxf(rm[v], x);
            rs[v] = rs[v] * __expf(rm[v] - m) + __expf(x - m);
            rm[v] = m;
        }
    }

#pragma unroll
    for (int v = 0; v < 8; ++v) {
        int lrow = mbase + v + 8 * laneHi;
        int slot = ni * 16 + ln;
        s_m[lrow][slot] = rm[v];
        s_s[lrow][slot] = rs[v];
        if (cap[v]) s_tgt[lrow] = tv[v];
    }
    __syncthreads();

    if (tid < 64) {
        float m = -INFINITY, s = 0.0f;
#pragma unroll 4
        for (int k = 0; k < 32; ++k) {
            float mk = s_m[tid][k], sk = s_s[tid][k];
            float mn = fmaxf(m, mk);
            s = s * __expf(m - mn) + sk * __expf(mk - mn);
            m = mn;
        }
        float lse = m + __logf(s);
        nll[rbase + tid] = lse - s_tgt[tid];
    }
}

__global__ void k_reduce(const float* __restrict__ nll, float* __restrict__ out) {
    __shared__ float sm[256];
    float s = 0.0f;
    for (int i = threadIdx.x; i < NROWS; i += 256) s += nll[i];
    sm[threadIdx.x] = s;
    __syncthreads();
    for (int st = 128; st > 0; st >>= 1) {
        if (threadIdx.x < st) sm[threadIdx.x] += sm[threadIdx.x + st];
        __syncthreads();
    }
    if (threadIdx.x == 0) out[0] = sm[0] / (float)NROWS;
}

// ---------------- launch ----------------

extern "C" void kernel_launch(void* const* d_in, const int* in_sizes, int n_in,
                              void* d_out, int out_size, void* d_ws, size_t ws_size,
                              hipStream_t stream) {
    const int*   input_data = (const int*)d_in[0];
    const int*   targets    = (const int*)d_in[1];
    const float* embedding  = (const float*)d_in[2];
    const float* W0         = (const float*)d_in[3];
    const float* b0         = (const float*)d_in[4];
    const float* W1         = (const float*)d_in[5];
    const float* b1         = (const float*)d_in[6];
    const float* softmax_w  = (const float*)d_in[7];
    const float* softmax_b  = (const float*)d_in[8];

    char* base = (char*)d_ws;
    size_t off = 0;
    auto take = [&](size_t bytes) -> void* {
        void* p = base + off;
        off += (bytes + 255) & ~(size_t)255;
        return p;
    };
    __bf16* W0t  = (__bf16*)take((size_t)K2 * G4 * 2);
    __bf16* W1t  = (__bf16*)take((size_t)K2 * G4 * 2);
    __bf16* SWt  = (__bf16*)take((size_t)V_ * U_ * 2);
    __bf16* Xbf  = (__bf16*)take((size_t)NROWS * U_ * 2);
    float*  Zx   = (float*)take((size_t)NROWS * G4 * 4);
    float*  c0   = (float*)take((size_t)B_ * U_ * 4);
    float*  c1   = (float*)take((size_t)B_ * U_ * 4);
    __bf16* h0b0 = (__bf16*)take((size_t)B_ * U_ * 2);
    __bf16* h0b1 = (__bf16*)take((size_t)B_ * U_ * 2);
    __bf16* h1b0 = (__bf16*)take((size_t)B_ * U_ * 2);
    __bf16* h1b1 = (__bf16*)take((size_t)B_ * U_ * 2);
    __bf16* outs = (__bf16*)take((size_t)NROWS * U_ * 2);
    float*  nll  = (float*)take((size_t)NROWS * 4);
    __bf16* h0buf[2] = { h0b0, h0b1 };
    __bf16* h1buf[2] = { h1b0, h1b1 };

    // weight conversion to K-major bf16
    k_f2bfT<<<(K2 * G4 + 255) / 256, 256, 0, stream>>>(W0, W0t, K2, G4);
    k_f2bfT<<<(K2 * G4 + 255) / 256, 256, 0, stream>>>(W1, W1t, K2, G4);
    k_f2bfT<<<(U_ * V_ + 255) / 256, 256, 0, stream>>>(softmax_w, SWt, U_, V_);

    // embedding gather (time-major), then hoisted x @ W0x + b0 for all timesteps
    k_embed<<<NROWS, U_, 0, stream>>>(input_data, embedding, Xbf);
    k_zx<<<dim3(NROWS / 64, G4 / 32), 256, 0, stream>>>(Xbf, W0t, b0, Zx);

    // zero recurrent state (fresh every call)
    hipMemsetAsync(c0, 0, (size_t)B_ * U_ * 4, stream);
    hipMemsetAsync(c1, 0, (size_t)B_ * U_ * 4, stream);
    hipMemsetAsync(h0b0, 0, (size_t)B_ * U_ * 2, stream);
    hipMemsetAsync(h0b1, 0, (size_t)B_ * U_ * 2, stream);
    hipMemsetAsync(h1b0, 0, (size_t)B_ * U_ * 2, stream);
    hipMemsetAsync(h1b1, 0, (size_t)B_ * U_ * 2, stream);

    // sequential recurrence: double-buffered h, in-place c
    for (int t = 0; t < T_; ++t) {
        k_lstm_l0<<<8, 256, 0, stream>>>(Zx + (size_t)t * B_ * G4,
                                         h0buf[(t + 1) & 1], W0t, c0, h0buf[t & 1]);
        k_lstm_l1<<<8, 256, 0, stream>>>(b1, h0buf[t & 1], h1buf[(t + 1) & 1],
                                         W1t, c1, h1buf[t & 1],
                                         outs + (size_t)t * B_ * U_);
    }

    // fused logits + online log-sum-exp + target gather, then scalar reduce
    k_proj_nll<<<NROWS / 64, 256, 0, stream>>>(outs, SWt, softmax_b, targets, nll);
    k_reduce<<<1, 256, 0, stream>>>(nll, (float*)d_out);
}
